// PatchEmbeddings_65352222376627
// MI455X (gfx1250) — compile-verified, matrix operand order checked by reading, not run on hardware
//
#include <hip/hip_runtime.h>

// ---- CDNA5 WMMA types ----
typedef __attribute__((ext_vector_type(16))) __bf16 v16bf;
typedef __attribute__((ext_vector_type(8)))  float  v8f;

union ABPack {
    uint4 u[2];   // 32 bytes
    v16bf v;      // 16 x bf16
};

// Problem constants
#define BATCH   32
#define IMG     512
#define CHN     3
#define PSIDE   16
#define GW      32          // grid of patches per side
#define NPATCH  1024        // GW*GW
#define KDIM    768         // P*P*C
#define NDIM    256         // hidden
#define ROWS    (BATCH * NPATCH)   // 32768 GEMM rows

// Tile sizes
#define BM 64
#define BK 32
#define BN 256
#define KITERS (KDIM / BK)  // 24

__global__ __launch_bounds__(256)
void PatchEmbeddings_gemm_kernel(const float* __restrict__ x,
                                 const float* __restrict__ Wm,
                                 const float* __restrict__ bias,
                                 const float* __restrict__ pos,
                                 float* __restrict__ out)
{
    // Double-buffered LDS tiles (bf16): A[m][k], B[n][k] (W transposed)
    __shared__ __align__(16) __bf16 lA[2][BM * BK];   // 2 x 4 KB
    __shared__ __align__(16) __bf16 lB[2][BN * BK];   // 2 x 16 KB

    const int tid     = threadIdx.x;
    const int wave    = tid >> 5;
    const int lane    = tid & 31;
    const int block_m = blockIdx.x * BM;

    const int mt = wave >> 1;     // 0..3 : M sub-tile
    const int nh = wave & 1;      // 0..1 : N half (128 cols)

    // WMMA 16-bit operand lane mapping:
    // lanes 0..15 -> row/col = lane, K in {0..7, 16..23}; lanes 16..31 -> K +8
    const int aklo = (lane >> 4) << 3;            // 0 or 8
    const int arow = mt * 16 + (lane & 15);       // A row within tile
    const int nl   = lane & 15;

    // Accumulators pre-loaded with bias (C/D layout: all 8 M-rows share n)
    v8f acc[8];
    #pragma unroll
    for (int nt = 0; nt < 8; ++nt) {
        float bv = bias[nh * 128 + nt * 16 + nl];
        v8f a0 = {bv, bv, bv, bv, bv, bv, bv, bv};
        acc[nt] = a0;
    }

    // Register staging for next tile's global data
    float  stA[8];
    float4 stW[8];

    // Issue global loads for K-chunk starting at kk
    auto issue_loads = [&](int kk) {
        #pragma unroll
        for (int i = 0; i < 8; ++i) {
            int e    = i * 256 + tid;            // 0..2047
            int mr   = e >> 5;                   // 0..63
            int kc   = e & 31;                   // 0..31
            int gm   = block_m + mr;
            int bb   = gm >> 10;
            int nidx = gm & (NPATCH - 1);
            int prow = nidx >> 5;
            int pcol = nidx & (GW - 1);
            int d    = kk + kc;
            int pr   = d / 48;
            int off  = d - pr * 48;
            int h    = prow * PSIDE + pr;
            long addr = ((long)(bb * IMG + h) * IMG + pcol * PSIDE) * CHN + off;
            stA[i] = x[addr];
        }
        #pragma unroll
        for (int i = 0; i < 8; ++i) {
            int e   = i * 256 + tid;
            int kr  = e >> 6;                    // 0..31
            int nc4 = e & 63;
            stW[i] = *(const float4*)(Wm + (long)(kk + kr) * NDIM + nc4 * 4);
        }
    };

    // Convert staged registers and store into LDS buffer `buf`
    auto store_tiles = [&](int buf) {
        #pragma unroll
        for (int i = 0; i < 8; ++i) {
            int e  = i * 256 + tid;
            int mr = e >> 5;
            int kc = e & 31;
            lA[buf][mr * BK + kc] = (__bf16)stA[i];
        }
        #pragma unroll
        for (int i = 0; i < 8; ++i) {
            int e  = i * 256 + tid;
            int kr = e >> 6;
            int n0 = (e & 63) * 4;
            lB[buf][(n0 + 0) * BK + kr] = (__bf16)stW[i].x;
            lB[buf][(n0 + 1) * BK + kr] = (__bf16)stW[i].y;
            lB[buf][(n0 + 2) * BK + kr] = (__bf16)stW[i].z;
            lB[buf][(n0 + 3) * BK + kr] = (__bf16)stW[i].w;
        }
    };

    // Prologue: fill buffer 0
    issue_loads(0);
    store_tiles(0);
    __syncthreads();

    for (int it = 0; it < KITERS; ++it) {
        const int cur  = it & 1;
        const bool more = (it + 1) < KITERS;

        // Overlap: start next tile's global loads before computing current tile
        if (more) issue_loads((it + 1) * BK);

        // A operand: two contiguous 16B LDS reads
        ABPack a;
        a.u[0] = *(const uint4*)&lA[cur][arow * BK + aklo];
        a.u[1] = *(const uint4*)&lA[cur][arow * BK + 16 + aklo];

        const int nbase = nh * 128;
        // Batch B-operand loads 4 tiles at a time, then 4 back-to-back WMMAs
        #pragma unroll
        for (int g = 0; g < 2; ++g) {
            ABPack bmx[4];
            #pragma unroll
            for (int t = 0; t < 4; ++t) {
                int n = nbase + (g * 4 + t) * 16 + nl;
                bmx[t].u[0] = *(const uint4*)&lB[cur][n * BK + aklo];
                bmx[t].u[1] = *(const uint4*)&lB[cur][n * BK + 16 + aklo];
            }
            #pragma unroll
            for (int t = 0; t < 4; ++t) {
                acc[g * 4 + t] = __builtin_amdgcn_wmma_f32_16x16x32_bf16(
                    false, a.v, false, bmx[t].v, (short)0, acc[g * 4 + t],
                    false, false);
            }
        }

        // Store next tile into the other buffer after compute has consumed LDS
        if (more) store_tiles(cur ^ 1);
        __syncthreads();
    }

    // ---------- Epilogue: + pos_embedding (bias already in acc) ----------
    // C/D layout: lanes 0..15 -> N=lane, M=vgpr; lanes 16..31 -> N=lane-16, M=8+vgpr
    const int gm_base = block_m + mt * 16 + ((lane >> 4) << 3);
    #pragma unroll
    for (int v = 0; v < 8; ++v) {
        int m    = gm_base + v;
        int bb   = m >> 10;
        int nidx = m & (NPATCH - 1);
        long rowo = ((long)bb * (NPATCH + 1) + 1 + nidx) * NDIM;
        const float* prow = pos + rowo;
        float*       orow = out + rowo;
        #pragma unroll
        for (int nt = 0; nt < 8; ++nt) {
            int n = nh * 128 + nt * 16 + nl;
            orow[n] = acc[nt][v] + prow[n];
        }
    }
}

// cls row: out[b, 0, :] = cls_token[b, :] + pos[b, 0, :]
__global__ __launch_bounds__(256)
void PatchEmbeddings_cls_kernel(const float* __restrict__ cls,
                                const float* __restrict__ pos,
                                float* __restrict__ out)
{
    int i  = blockIdx.x * 256 + threadIdx.x;   // 0..8191
    int bb = i >> 8;
    int h  = i & 255;
    long o = (long)bb * (NPATCH + 1) * NDIM + h;
    out[o] = cls[i] + pos[o];
}

extern "C" void kernel_launch(void* const* d_in, const int* in_sizes, int n_in,
                              void* d_out, int out_size, void* d_ws, size_t ws_size,
                              hipStream_t stream) {
    const float* x    = (const float*)d_in[0];   // (32,512,512,3)
    const float* Wm   = (const float*)d_in[1];   // (768,256)
    const float* bias = (const float*)d_in[2];   // (256,)
    const float* cls  = (const float*)d_in[3];   // (32,1,256)
    const float* pos  = (const float*)d_in[4];   // (32,1025,256)
    float* out = (float*)d_out;                  // (32,1025,256)

    PatchEmbeddings_gemm_kernel<<<dim3(ROWS / BM), dim3(256), 0, stream>>>(
        x, Wm, bias, pos, out);
    PatchEmbeddings_cls_kernel<<<dim3((BATCH * NDIM) / 256), dim3(256), 0, stream>>>(
        cls, pos, out);
}
